// VectorQuantizerEMA_65695819760355
// MI455X (gfx1250) — matrix operand order repaired and checked by reading, not compiled, not run on hardware
//
#include <hip/hip_runtime.h>

// MI455X / gfx1250 VQ nearest-neighbor quantizer.
// dist[m][n] = ||e_n||^2 - 2 * (z_m . e_n); argmin over n; gather e.
// GEMM (131072 x 1024 x 64) done with v_wmma_f32_16x16x32_f16 using a
// hi/lo split-f16 3-term product for ~fp32-accurate scores.

typedef __attribute__((ext_vector_type(16))) _Float16 v16h;
typedef __attribute__((ext_vector_type(8)))  float    v8f;

#define K_CODES     1024
#define C_DIM       64
#define HW_DIM      4096              // 64*64 pixels per batch image
#define TILES       (K_CODES / 16)    // 64 code tiles of 16
#define CHUNK_TILES 8
#define NCHUNKS     (TILES / CHUNK_TILES)
#define PIX_PER_WG  128               // 8 waves * 16 pixels

__global__ __launch_bounds__(256) void vq_argmin_kernel(
    const float* __restrict__ z,      // [32,64,64,64] = [B,C,H,W]
    const float* __restrict__ emb,    // [1024,64]
    float* __restrict__ out)          // [32,64,64,64]
{
  // LDS: B fragments (f16 hi/lo, WMMA B layout) for one 8-tile chunk + code norms.
  __shared__ __attribute__((aligned(32))) _Float16 sBh[CHUNK_TILES * 2 * 32 * 16]; // 16 KB
  __shared__ __attribute__((aligned(32))) _Float16 sBl[CHUNK_TILES * 2 * 32 * 16]; // 16 KB
  __shared__ float sNorm[K_CODES];                                                 // 4 KB

  const int tid  = threadIdx.x;
  const int lane = tid & 31;
  const int wave = tid >> 5;
  const int half = lane >> 4;   // which 16-lane half of the wave32
  const int nl   = lane & 15;

  // ---------- exact fp32 codebook norms ----------
  for (int n = tid; n < K_CODES; n += 256) {
    const float* er = emb + n * C_DIM;
    float s = 0.f;
#pragma unroll
    for (int c = 0; c < C_DIM; ++c) s += er[c] * er[c];
    sNorm[n] = s;
  }

  // ---------- build A fragments: 16 pixels x 64 channels, f16 hi/lo ----------
  // ISA 16-bit A 16x32 layout: lane M = lane&15; element e -> K =
  //   kblock*32 + (j>=4?16:0) + (lane>>4)*8 + (j&3)*2 + (e&1), j = e>>1.
  const int p     = blockIdx.x * PIX_PER_WG + wave * 16 + nl;   // this lane's M-row pixel
  const long zoff = (long)(p >> 12) * (C_DIM * HW_DIM) + (p & (HW_DIM - 1));
  v16h Ah0, Ah1, Al0, Al1;
#pragma unroll
  for (int e = 0; e < 16; ++e) {
    const int j  = e >> 1;
    const int kb = ((j >= 4) ? 16 : 0) + half * 8 + (j & 3) * 2 + (e & 1);
    {
      float x = z[zoff + (long)(kb) * HW_DIM];          // k-block 0: channels 0..31
      _Float16 h = (_Float16)x;
      Ah0[e] = h;  Al0[e] = (_Float16)(x - (float)h);
    }
    {
      float x = z[zoff + (long)(kb + 32) * HW_DIM];     // k-block 1: channels 32..63
      _Float16 h = (_Float16)x;
      Ah1[e] = h;  Al1[e] = (_Float16)(x - (float)h);
    }
  }

  float bestv[8];
  int   besti[8];
#pragma unroll
  for (int r = 0; r < 8; ++r) { bestv[r] = 3.4e38f; besti[r] = 0; }

  // ---------- sweep the codebook in 8-tile chunks staged in LDS ----------
  for (int ch = 0; ch < NCHUNKS; ++ch) {
    __syncthreads();  // previous chunk fully consumed (also covers sNorm on ch==0)

    // Convert emb fp32 -> f16 hi/lo directly into WMMA B layout.
    // B element e of lane ln for (tile, kblock): holds emb[n][K],
    //   n = tile*16 + (ln&15), K = kblock*32 + (e>=8?16:0) + (ln>>4)*8 + (e&7).
    for (int q = tid; q < CHUNK_TILES * 2 * 32; q += 256) {
      const int tl  = q >> 6;
      const int kb  = (q >> 5) & 1;
      const int ln  = q & 31;
      const int n   = (ch * CHUNK_TILES + tl) * 16 + (ln & 15);
      const int kst = kb * 32 + (ln >> 4) * 8;
      const float* src = emb + n * C_DIM + kst;
      const int dst = q * 16;
#pragma unroll
      for (int e = 0; e < 16; ++e) {
        float x = src[e < 8 ? e : e + 8];
        _Float16 h = (_Float16)x;
        sBh[dst + e] = h;
        sBl[dst + e] = (_Float16)(x - (float)h);
      }
    }
    __syncthreads();

    for (int tl = 0; tl < CHUNK_TILES; ++tl) {
      const int t = ch * CHUNK_TILES + tl;
      const v16h Bh0 = *(const v16h*)&sBh[((tl * 2 + 0) * 32 + lane) * 16];
      const v16h Bl0 = *(const v16h*)&sBl[((tl * 2 + 0) * 32 + lane) * 16];
      const v16h Bh1 = *(const v16h*)&sBh[((tl * 2 + 1) * 32 + lane) * 16];
      const v16h Bl1 = *(const v16h*)&sBl[((tl * 2 + 1) * 32 + lane) * 16];

      // S = Ah*Bh + Ah*Bl + Al*Bh  (drop Al*Bl), fp32 accumulate.
      v8f acc = {};
      acc = __builtin_amdgcn_wmma_f32_16x16x32_f16(false, Ah0, false, Bh0, (short)0, acc, false, false);
      acc = __builtin_amdgcn_wmma_f32_16x16x32_f16(false, Ah0, false, Bl0, (short)0, acc, false, false);
      acc = __builtin_amdgcn_wmma_f32_16x16x32_f16(false, Al0, false, Bh0, (short)0, acc, false, false);
      acc = __builtin_amdgcn_wmma_f32_16x16x32_f16(false, Ah1, false, Bh1, (short)0, acc, false, false);
      acc = __builtin_amdgcn_wmma_f32_16x16x32_f16(false, Ah1, false, Bl1, (short)0, acc, false, false);
      acc = __builtin_amdgcn_wmma_f32_16x16x32_f16(false, Al1, false, Bh1, (short)0, acc, false, false);

      const float nrm  = sNorm[t * 16 + nl];
      const int   nidx = t * 16 + nl;
#pragma unroll
      for (int r = 0; r < 8; ++r) {
        // D layout: VGPR r = row (r + half*8), column nl of this tile.
        const float d = nrm - 2.0f * acc[r];
        if (d < bestv[r]) { bestv[r] = d; besti[r] = nidx; }
      }
    }
  }

  // ---------- reduce argmin across the 16 lanes of each half ----------
#pragma unroll
  for (int r = 0; r < 8; ++r) {
    float v = bestv[r];
    int   i = besti[r];
#pragma unroll
    for (int m = 1; m <= 8; m <<= 1) {
      const float ov = __shfl_xor(v, m, 32);
      const int   oi = __shfl_xor(i, m, 32);
      if (ov < v || (ov == v && oi < i)) { v = ov; i = oi; }  // first-min tiebreak
    }
    besti[r] = i;   // all lanes of the half now hold row (r + half*8)'s winner
  }
  int otheri[8];
#pragma unroll
  for (int r = 0; r < 8; ++r) otheri[r] = __shfl_xor(besti[r], 16, 32);

  // ---------- gather winning embedding rows, scatter to [B,C,H,W] ----------
  const int mi = nl;                         // lane writes pixel (base + mi)
  const int r2 = mi & 7;
  const int codeIdx = ((mi >> 3) == half) ? besti[r2] : otheri[r2];
  const long obase  = (long)(p >> 12) * (C_DIM * HW_DIM) + (p & (HW_DIM - 1));
  const float* erow = emb + codeIdx * C_DIM;
#pragma unroll
  for (int it = 0; it < 32; ++it) {
    const int c = it + 32 * half;            // lanes 0-15: c=0..31, lanes 16-31: c=32..63
    out[obase + (long)c * HW_DIM] = erow[c];
  }
}

extern "C" void kernel_launch(void* const* d_in, const int* in_sizes, int n_in,
                              void* d_out, int out_size, void* d_ws, size_t ws_size,
                              hipStream_t stream) {
  const float* z   = (const float*)d_in[0];   // z_e  [32,64,64,64] fp32
  const float* emb = (const float*)d_in[1];   // embedding [1024,64] fp32
  float* out = (float*)d_out;                 // [32,64,64,64] fp32

  const int n_pix = in_sizes[0] / C_DIM;      // 131072
  const int grid  = n_pix / PIX_PER_WG;       // 1024 workgroups of 256 threads
  vq_argmin_kernel<<<grid, 256, 0, stream>>>(z, emb, out);
}
